// myYOLOv2_60266981097812
// MI455X (gfx1250) — matrix operand order repaired and checked by reading, not compiled, not run on hardware
//
#include <hip/hip_runtime.h>

// ---------------- CDNA5 WMMA types ----------------
typedef __attribute__((ext_vector_type(16))) _Float16 v16h;
typedef __attribute__((ext_vector_type(8)))  _Float16 v8h;
typedef __attribute__((ext_vector_type(8)))  float    v8f;

union HF { v8h h[2]; v16h v; };

__device__ __forceinline__ float lrelu(float x){ return x > 0.0f ? x : 0.1f * x; }
__device__ __forceinline__ float sigmoidf(float x){ return 1.0f / (1.0f + __expf(-x)); }

__device__ __forceinline__ v8f wmma_f16(v16h a, v16h b, v8f c){
  return __builtin_amdgcn_wmma_f32_16x16x32_f16(false, a, false, b, (short)0, c, false, false);
}

// A-matrix fragment (16x32 f16, M x K, row-major weights).
// lane L: row m=L%16, khalf=L>>4; element j -> k=(j<8?0:16)+khalf*8+(j%8):
// two contiguous 16B chunks -> global_load_b128 pairs.
__device__ __forceinline__ v16h fragA(const _Float16* __restrict__ A, int K, int mrow, int k0, int lane){
  int kh = lane >> 4;
  const _Float16* b = A + (size_t)mrow * K + k0 + kh * 8;
  HF f;
  f.h[0] = *(const v8h*)(b);
  f.h[1] = *(const v8h*)(b + 16);
  return f.v;
}

// B-matrix fragment (32x16 f16, K x N). Activations NHWC (channel-contiguous):
// lane L (col n=L%16, khalf=L>>4) reads k=k0+(j<8?0:16)+khalf*8+(j%8) (same order as fragA).
__device__ __forceinline__ v16h fragB(const _Float16* __restrict__ act, int C, int prow, int k0, int lane){
  int kh = lane >> 4;
  const _Float16* b = act + (size_t)prow * C + k0 + kh * 8;
  HF f;
  f.h[0] = *(const v8h*)(b);
  f.h[1] = *(const v8h*)(b + 16);
  return f.v;
}

// ---------------- fused epilogues ----------------
enum { EPI_PAD_HALF = 0, EPI_REORG = 1, EPI_HALF = 2, EPI_PRED = 3 };

template<int EPI>
__device__ __forceinline__ void store_out(float v, int co, int p,
                                          _Float16* outh, float* outf,
                                          int cbase, int cstride){
  if constexpr (EPI == EPI_PAD_HALF){           // lrelu -> padded 40x40 NHWC
    int row = (p / 38 + 1) * 40 + (p % 38 + 1);
    outh[(size_t)row * cstride + cbase + co] = (_Float16)lrelu(v);
  } else if constexpr (EPI == EPI_REORG){       // lrelu + reorg s=2 into concat buf (ch 0..127)
    int Y = p / 76, X = p % 76;
    int row = ((Y >> 1) + 1) * 40 + ((X >> 1) + 1);
    int ch  = ((Y & 1) * 2 + (X & 1)) * 32 + co;
    outh[(size_t)row * 640 + ch] = (_Float16)lrelu(v);
  } else if constexpr (EPI == EPI_HALF){        // lrelu -> plain NHWC
    outh[(size_t)p * cstride + co] = (_Float16)lrelu(v);
  } else {                                      // raw f32 prediction head
    outf[(size_t)p * 128 + co] = v;
  }
}

// store one 16x16 C/D tile (VGPR d -> m = d + (lane>=16)*8, n = lane%16)
template<int EPI>
__device__ __forceinline__ void store_tile(const v8f& acc, int m0, int n0, int N, int lane,
                                           const float* __restrict__ bias,
                                           _Float16* outh, float* outf, int cbase, int cstride){
  int n = lane & 15, pp = n0 + n;
  if (pp >= N) return;
  #pragma unroll
  for (int d = 0; d < 8; ++d){
    int co = m0 + d + ((lane >> 4) << 3);
    store_out<EPI>(acc[d] + bias[co], co, pp, outh, outf, cbase, cstride);
  }
}

// ---------------- 1x1 conv = GEMM, 32x32 tile per wave (2x2 WMMA blocking) ----------------
template<int EPI>
__global__ void __launch_bounds__(256)
k_gemm1x1(const _Float16* __restrict__ A, const _Float16* __restrict__ B,
          const float* __restrict__ bias, int M, int N, int K,
          _Float16* outh, float* outf, int cbase, int cstride)
{
  int lane  = threadIdx.x & 31;
  int wave  = threadIdx.x >> 5;
  int ntile = blockIdx.x * (blockDim.x >> 5) + wave;
  int ntiles = (N + 31) >> 5;
  if (ntile >= ntiles) return;                 // wave-uniform exit
  int m0 = blockIdx.y * 32, n0 = ntile * 32;

  int mr0 = m0 + (lane & 15), mr1 = mr0 + 16;              // M multiple of 32
  int p0 = n0 + (lane & 15);      if (p0 >= N) p0 = N - 1; // clamp: EXEC stays full
  int p1 = n0 + 16 + (lane & 15); if (p1 >= N) p1 = N - 1;

  v8f a00 = {}, a01 = {}, a10 = {}, a11 = {};
  for (int k0 = 0; k0 < K; k0 += 32){
    if (k0 + 32 < K){
      __builtin_prefetch(A + (size_t)mr0 * K + k0 + 32, 0, 1);
      __builtin_prefetch(B + (size_t)p0 * K + k0 + 32, 0, 1);
    }
    v16h fa0 = fragA(A, K, mr0, k0, lane);
    v16h fa1 = fragA(A, K, mr1, k0, lane);
    v16h fb0 = fragB(B, K, p0, k0, lane);
    v16h fb1 = fragB(B, K, p1, k0, lane);
    a00 = wmma_f16(fa0, fb0, a00);
    a01 = wmma_f16(fa0, fb1, a01);
    a10 = wmma_f16(fa1, fb0, a10);
    a11 = wmma_f16(fa1, fb1, a11);
  }
  store_tile<EPI>(a00, m0,      n0,      N, lane, bias, outh, outf, cbase, cstride);
  store_tile<EPI>(a01, m0,      n0 + 16, N, lane, bias, outh, outf, cbase, cstride);
  store_tile<EPI>(a10, m0 + 16, n0,      N, lane, bias, outh, outf, cbase, cstride);
  store_tile<EPI>(a11, m0 + 16, n0 + 16, N, lane, bias, outh, outf, cbase, cstride);
}

// ---------------- 3x3 conv = 9 accumulated GEMM slices, 32x32 tile per wave ----------------
template<int EPI>
__global__ void __launch_bounds__(256)
k_gemm3x3(const _Float16* __restrict__ Wperm,   // [9][M][Cin]
          const _Float16* __restrict__ actpad,  // [40*40][Cin], zero border
          const float* __restrict__ bias, int M, int Cin,
          _Float16* outh, float* outf, int cbase, int cstride)
{
  const int N = 38 * 38;
  int lane  = threadIdx.x & 31;
  int wave  = threadIdx.x >> 5;
  int ntile = blockIdx.x * (blockDim.x >> 5) + wave;
  int ntiles = (N + 31) >> 5;
  if (ntile >= ntiles) return;
  int m0 = blockIdx.y * 32, n0 = ntile * 32;

  int mr0 = m0 + (lane & 15), mr1 = mr0 + 16;
  int p0 = n0 + (lane & 15);      if (p0 >= N) p0 = N - 1;
  int p1 = n0 + 16 + (lane & 15); if (p1 >= N) p1 = N - 1;
  int y0 = p0 / 38, x0 = p0 % 38;
  int y1 = p1 / 38, x1 = p1 % 38;

  v8f a00 = {}, a01 = {}, a10 = {}, a11 = {};
  for (int t = 0; t < 9; ++t){
    int dy = t / 3, dx = t % 3;
    int pr0 = (y0 + dy) * 40 + (x0 + dx);
    int pr1 = (y1 + dy) * 40 + (x1 + dx);
    const _Float16* As = Wperm + (size_t)t * M * Cin;
    for (int k0 = 0; k0 < Cin; k0 += 32){
      if (k0 + 32 < Cin){
        __builtin_prefetch(As + (size_t)mr0 * Cin + k0 + 32, 0, 1);
        __builtin_prefetch(actpad + (size_t)pr0 * Cin + k0 + 32, 0, 1);
      }
      v16h fa0 = fragA(As, Cin, mr0, k0, lane);
      v16h fa1 = fragA(As, Cin, mr1, k0, lane);
      v16h fb0 = fragB(actpad, Cin, pr0, k0, lane);
      v16h fb1 = fragB(actpad, Cin, pr1, k0, lane);
      a00 = wmma_f16(fa0, fb0, a00);
      a01 = wmma_f16(fa0, fb1, a01);
      a10 = wmma_f16(fa1, fb0, a10);
      a11 = wmma_f16(fa1, fb1, a11);
    }
  }
  store_tile<EPI>(a00, m0,      n0,      N, lane, bias, outh, outf, cbase, cstride);
  store_tile<EPI>(a01, m0,      n0 + 16, N, lane, bias, outh, outf, cbase, cstride);
  store_tile<EPI>(a10, m0 + 16, n0,      N, lane, bias, outh, outf, cbase, cstride);
  store_tile<EPI>(a11, m0 + 16, n0 + 16, N, lane, bias, outh, outf, cbase, cstride);
}

// ---------------- conversion / layout kernels ----------------
__global__ void k_f32_to_f16(const float* __restrict__ in, _Float16* __restrict__ out, int n){
  int i = blockIdx.x * blockDim.x + threadIdx.x;
  if (i < n) out[i] = (_Float16)in[i];
}
// OIHW -> [t=dy*3+dx][O][I] f16
__global__ void k_perm3x3(const float* __restrict__ in, _Float16* __restrict__ out, int O, int I){
  int i = blockIdx.x * blockDim.x + threadIdx.x;
  int n = O * I * 9;
  if (i >= n) return;
  int t = i / (O * I), rem = i % (O * I);
  int o = rem / I, ci = rem % I;
  out[i] = (_Float16)in[(size_t)(o * I + ci) * 9 + t];
}
// NCHW f32 -> pixel-major NHWC f16
__global__ void k_nchw_to_nhwc(const float* __restrict__ in, _Float16* __restrict__ out, int C, int HW){
  int i = blockIdx.x * blockDim.x + threadIdx.x;
  if (i >= C * HW) return;
  int p = i / C, c = i % C;
  out[i] = (_Float16)in[(size_t)c * HW + p];
}
// Wp (125x512) -> padded 128x512 f16, and bp -> padded 128 f32
__global__ void k_pad_wp(const float* __restrict__ Wp, const float* __restrict__ bp,
                         _Float16* __restrict__ Wph, float* __restrict__ bpad){
  int i = blockIdx.x * blockDim.x + threadIdx.x;
  if (i >= 128 * 512) return;
  int co = i / 512, ci = i % 512;
  Wph[i] = (co < 125) ? (_Float16)Wp[(size_t)co * 512 + ci] : (_Float16)0.0f;
  if (i < 128) bpad[i] = (i < 125) ? bp[i] : 0.0f;
}
__global__ void k_fill_h0(_Float16* __restrict__ p, int n){
  int i = blockIdx.x * blockDim.x + threadIdx.x;
  if (i < n) p[i] = (_Float16)0.0f;
}

// ---------------- box decode ----------------
__global__ void k_decode(const float* __restrict__ pred /* [1444][128] */,
                         float* __restrict__ boxes, float* __restrict__ scores,
                         int* __restrict__ clsi, int* __restrict__ valid){
  const float AW[5] = {1.19f, 2.79f, 4.53f, 8.06f, 10.32f};
  const float AH[5] = {1.98f, 4.59f, 8.92f, 5.29f, 10.65f};
  int i = blockIdx.x * blockDim.x + threadIdx.x;
  if (i >= 7220) return;
  int p = i / 5, a = i % 5;
  const float* pr = pred + (size_t)p * 128;

  float obj = sigmoidf(pr[a]);
  const float* c = pr + 5 + a * 20;
  float mx = c[0]; int am = 0;
  #pragma unroll
  for (int j = 1; j < 20; ++j){ float v = c[j]; if (v > mx){ mx = v; am = j; } }
  float s = 0.0f;
  #pragma unroll
  for (int j = 0; j < 20; ++j) s += __expf(c[j] - mx);
  float score = obj / s;               // max softmax prob = 1/s

  const float* tb = pr + 105 + a * 4;
  float gx = (float)(p % 38), gy = (float)(p / 38);
  float cx = sigmoidf(tb[0]) + gx;
  float cy = sigmoidf(tb[1]) + gy;
  float bw = __expf(tb[2]) * AW[a];
  float bh = __expf(tb[3]) * AH[a];
  const float inv = 32.0f / 1216.0f;   // STRIDE / IN_W (normalized boxes)
  boxes[i * 4 + 0] = (cx - 0.5f * bw) * inv;
  boxes[i * 4 + 1] = (cy - 0.5f * bh) * inv;
  boxes[i * 4 + 2] = (cx + 0.5f * bw) * inv;
  boxes[i * 4 + 3] = (cy + 0.5f * bh) * inv;
  scores[i] = score;
  clsi[i]   = am;
  valid[i]  = (score >= 0.01f) ? 1 : 0;
}

// ---------------- single-block LDS bitonic argsort (descending) ----------------
__global__ void __launch_bounds__(1024) k_sort(const float* __restrict__ scores, int* __restrict__ order){
  const int NS = 8192;
  __shared__ float skey[8192];
  __shared__ int   sidx[8192];
  int t = threadIdx.x;
  for (int i = t; i < NS; i += 1024){
    skey[i] = (i < 7220) ? scores[i] : -1e30f;
    sidx[i] = i;
  }
  __syncthreads();
  for (int ksz = 2; ksz <= NS; ksz <<= 1){
    for (int j = ksz >> 1; j > 0; j >>= 1){
      for (int i = t; i < NS; i += 1024){
        int q = i ^ j;
        if (q > i){
          bool dir = ((i & ksz) == 0);           // descending comparator
          float a = skey[i], b = skey[q];
          if ((a < b) == dir){
            skey[i] = b; skey[q] = a;
            int ti = sidx[i]; sidx[i] = sidx[q]; sidx[q] = ti;
          }
        }
      }
      __syncthreads();
    }
  }
  for (int i = t; i < NS; i += 1024) order[i] = sidx[i];
}

// ---------------- greedy per-class NMS (sequential over sorted order) ----------------
__global__ void __launch_bounds__(512)
k_nms(const float* __restrict__ boxes, const int* __restrict__ clsi, const int* __restrict__ valid,
      const int* __restrict__ order, int* __restrict__ keep_orig){
  const int N = 7220;
  __shared__ unsigned char keep[7220];
  __shared__ float bx1, by1, bx2, by2;
  __shared__ int bcls, bval, supp;
  int t = threadIdx.x;
  for (int i = t; i < N; i += blockDim.x) keep[i] = 0;
  __syncthreads();
  for (int i = 0; i < N; ++i){
    if (t == 0){
      int oi = order[i];
      bx1 = boxes[oi * 4 + 0]; by1 = boxes[oi * 4 + 1];
      bx2 = boxes[oi * 4 + 2]; by2 = boxes[oi * 4 + 3];
      bcls = clsi[oi]; bval = valid[oi]; supp = 0;
    }
    __syncthreads();
    float ax1 = bx1, ay1 = by1, ax2 = bx2, ay2 = by2;
    float areaA = (ax2 - ax1) * (ay2 - ay1);
    int mycls = bcls;
    int found = 0;
    for (int j = t; j < i; j += blockDim.x){
      if (!keep[j]) continue;
      int oj = order[j];
      if (clsi[oj] != mycls) continue;
      float cx1 = boxes[oj * 4 + 0], cy1 = boxes[oj * 4 + 1];
      float cx2 = boxes[oj * 4 + 2], cy2 = boxes[oj * 4 + 3];
      float iw = fmaxf(1e-28f, fminf(ax2, cx2) - fmaxf(ax1, cx1));
      float ih = fmaxf(1e-28f, fminf(ay2, cy2) - fmaxf(ay1, cy1));
      float inter = iw * ih;
      float areaC = (cx2 - cx1) * (cy2 - cy1);
      float iou = inter / (areaA + areaC - inter);
      if (iou > 0.5f){ found = 1; break; }
    }
    if (found) supp = 1;                       // benign race: all writers store 1
    __syncthreads();
    if (t == 0) keep[i] = (bval && !supp) ? 1 : 0;
    __syncthreads();
  }
  for (int i = t; i < N; i += blockDim.x) keep_orig[order[i]] = keep[i];
}

// ---------------- final outputs (bboxes, scores, cls, keep) ----------------
__global__ void k_finalize(const float* __restrict__ boxes, const float* __restrict__ scores,
                           const int* __restrict__ clsi, const int* __restrict__ keep,
                           float* __restrict__ out){
  int i = blockIdx.x * blockDim.x + threadIdx.x;
  if (i >= 7220) return;
  int k = keep[i];
  #pragma unroll
  for (int t = 0; t < 4; ++t){
    float v = boxes[i * 4 + t] * 1216.0f;
    v = fminf(fmaxf(v, 0.0f), 1215.0f) * (1.0f / 1216.0f);
    out[i * 4 + t] = k ? v : 0.0f;
  }
  out[28880 + i] = k ? scores[i] : 0.0f;
  int* co = (int*)(out + 36100);
  co[i] = k ? clsi[i] : -1;
  int* ko = (int*)(out + 43320);
  ko[i] = k;
}

// ---------------- host orchestration ----------------
extern "C" void kernel_launch(void* const* d_in, const int* in_sizes, int n_in,
                              void* d_out, int out_size, void* d_ws, size_t ws_size,
                              hipStream_t stream){
  (void)in_sizes; (void)n_in; (void)out_size; (void)ws_size;
  const float* fp_1 = (const float*)d_in[0];   // (1,256,76,76)
  const float* fp_2 = (const float*)d_in[1];   // (1,512,38,38)
  const float* W1 = (const float*)d_in[2];  const float* b1 = (const float*)d_in[3];
  const float* W2 = (const float*)d_in[4];  const float* b2 = (const float*)d_in[5];
  const float* Wr = (const float*)d_in[6];  const float* br = (const float*)d_in[7];
  const float* W3 = (const float*)d_in[8];  const float* b3 = (const float*)d_in[9];
  const float* Wp = (const float*)d_in[10]; const float* bp = (const float*)d_in[11];

  char* wsp = (char*)d_ws;
  auto alloc = [&](size_t bytes) -> void* {
    void* p = (void*)wsp;
    wsp += (bytes + 255) & ~(size_t)255;
    return p;
  };
  _Float16* W1h  = (_Float16*)alloc(256 * 512 * 2);
  _Float16* W2h  = (_Float16*)alloc((size_t)9 * 512 * 256 * 2);
  _Float16* Wrh  = (_Float16*)alloc(32 * 256 * 2);
  _Float16* W3h  = (_Float16*)alloc((size_t)9 * 512 * 640 * 2);
  _Float16* Wph  = (_Float16*)alloc(128 * 512 * 2);
  float*    bpad = (float*)alloc(128 * 4);
  _Float16* fp2h = (_Float16*)alloc((size_t)1444 * 512 * 2);
  _Float16* fp1h = (_Float16*)alloc((size_t)5776 * 256 * 2);
  _Float16* x1p  = (_Float16*)alloc((size_t)1600 * 256 * 2);   // padded 40x40 NHWC
  _Float16* cat  = (_Float16*)alloc((size_t)1600 * 640 * 2);   // padded concat buf
  _Float16* fph  = (_Float16*)alloc((size_t)1444 * 512 * 2);
  float*    pred = (float*)alloc((size_t)1444 * 128 * 4);
  float*    boxes  = (float*)alloc(7220 * 4 * 4);
  float*    scores = (float*)alloc(7220 * 4);
  int*      clsi   = (int*)alloc(7220 * 4);
  int*      valid  = (int*)alloc(7220 * 4);
  int*      order  = (int*)alloc(8192 * 4);
  int*      keepo  = (int*)alloc(7220 * 4);

  auto cdiv = [](int a, int b){ return (a + b - 1) / b; };

  // layout conversions (f32 -> f16, permutes, zero padded borders)
  k_f32_to_f16 <<<cdiv(256 * 512, 256), 256, 0, stream>>>(W1, W1h, 256 * 512);
  k_perm3x3    <<<cdiv(9 * 512 * 256, 256), 256, 0, stream>>>(W2, W2h, 512, 256);
  k_f32_to_f16 <<<cdiv(32 * 256, 256), 256, 0, stream>>>(Wr, Wrh, 32 * 256);
  k_perm3x3    <<<cdiv(9 * 512 * 640, 256), 256, 0, stream>>>(W3, W3h, 512, 640);
  k_pad_wp     <<<cdiv(128 * 512, 256), 256, 0, stream>>>(Wp, bp, Wph, bpad);
  k_nchw_to_nhwc<<<cdiv(512 * 1444, 256), 256, 0, stream>>>(fp_2, fp2h, 512, 1444);
  k_nchw_to_nhwc<<<cdiv(256 * 5776, 256), 256, 0, stream>>>(fp_1, fp1h, 256, 5776);
  k_fill_h0    <<<cdiv(1600 * 256, 256), 256, 0, stream>>>(x1p, 1600 * 256);
  k_fill_h0    <<<cdiv(1600 * 640, 256), 256, 0, stream>>>(cat, 1600 * 640);

  // tile counts for 32x32 waves-tiles (8 waves per 256-thread block)
  const int nt38 = cdiv(1444, 32);   // 46
  const int nt76 = cdiv(5776, 32);   // 181

  // conv1: 1x1 512->256 over 38x38, lrelu, into padded buffer
  k_gemm1x1<EPI_PAD_HALF><<<dim3(cdiv(nt38, 8), 256 / 32), 256, 0, stream>>>(
      W1h, fp2h, b1, 256, 1444, 512, x1p, nullptr, 0, 256);
  // convr: 1x1 256->32 over 76x76, lrelu + reorg -> cat channels [0,128)
  k_gemm1x1<EPI_REORG><<<dim3(cdiv(nt76, 8), 32 / 32), 256, 0, stream>>>(
      Wrh, fp1h, br, 32, 5776, 256, cat, nullptr, 0, 640);
  // conv2: 3x3 256->512, lrelu -> cat channels [128,640)
  k_gemm3x3<EPI_PAD_HALF><<<dim3(cdiv(nt38, 8), 512 / 32), 256, 0, stream>>>(
      W2h, x1p, b2, 512, 256, cat, nullptr, 128, 640);
  // conv3: 3x3 640->512, lrelu -> fph (plain NHWC)
  k_gemm3x3<EPI_HALF><<<dim3(cdiv(nt38, 8), 512 / 32), 256, 0, stream>>>(
      W3h, cat, b3, 512, 640, fph, nullptr, 0, 512);
  // convp: 1x1 512->125 (padded to 128), raw f32 predictions
  k_gemm1x1<EPI_PRED><<<dim3(cdiv(nt38, 8), 128 / 32), 256, 0, stream>>>(
      Wph, fph, bpad, 128, 1444, 512, nullptr, pred, 0, 128);

  // decode -> sort -> NMS -> outputs
  k_decode  <<<cdiv(7220, 256), 256, 0, stream>>>(pred, boxes, scores, clsi, valid);
  k_sort    <<<1, 1024, 0, stream>>>(scores, order);
  k_nms     <<<1, 512, 0, stream>>>(boxes, clsi, valid, order, keepo);
  k_finalize<<<cdiv(7220, 256), 256, 0, stream>>>(boxes, scores, clsi, keepo, (float*)d_out);
}
